// DOMTransformerLayer_14130442404236
// MI455X (gfx1250) — matrix operand order repaired and checked by reading, not compiled, not run on hardware
//
#include <hip/hip_runtime.h>
#include <math.h>

#define NT   6144
#define DM   256
#define NH   4
#define HD   64
#define DFF  1024

typedef __attribute__((ext_vector_type(16))) _Float16 v16h;
typedef __attribute__((ext_vector_type(8)))  _Float16 v8h;
typedef __attribute__((ext_vector_type(8)))  float    v8f;

// ---------------- WMMA fragment loaders (CDNA5 wave32 layouts, ISA 7.12.2) ----

// A fragment: 16x32 f16, row-major source with leading dim ld (halfs).
// lane l: row = l&15; K runs [kb, kb+8) and [kb+16, kb+24) with kb = (l>>4)*8.
__device__ __forceinline__ v16h load_frag_a(const _Float16* p, int ld, int lane) {
  int row = lane & 15;
  int kb  = (lane >> 4) * 8;
  const _Float16* base = p + row * ld + kb;
  v8h lo = *(const v8h*)(base);
  v8h hi = *(const v8h*)(base + 16);
  v16h r;
#pragma unroll
  for (int i = 0; i < 8; ++i) { r[i] = lo[i]; r[i + 8] = hi[i]; }
  return r;
}

// B fragment: 32x16 f16 (KxN) from an N-major ("Bt") source: Bt[n*ld + k].
// lane l: n = l&15; K runs [kb, kb+8) and [kb+16, kb+24) with kb = (l>>4)*8.
__device__ __forceinline__ v16h load_frag_b(const _Float16* bt, int ld, int lane) {
  int n  = lane & 15;
  int kb = (lane >> 4) * 8;
  const _Float16* base = bt + n * ld + kb;
  v8h lo = *(const v8h*)(base);
  v8h hi = *(const v8h*)(base + 16);
  v16h r;
#pragma unroll
  for (int i = 0; i < 8; ++i) { r[i] = lo[i]; r[i + 8] = hi[i]; }
  return r;
}

__device__ __forceinline__ v8f wmma_f16(v16h a, v16h b, v8f c) {
  return __builtin_amdgcn_wmma_f32_16x16x32_f16(false, a, false, b, (short)0, c,
                                                false, false);
}

// 16x64 output tile: one A fragment reused across 4 B fragments per K-step.
__device__ __forceinline__ void gemm_core4(const _Float16* A, const _Float16* Bt,
                                           int K, int lane, v8f acc[4]) {
  for (int k0 = 0; k0 < K; k0 += 32) {
    v16h a = load_frag_a(A + k0, K, lane);
#pragma unroll
    for (int s = 0; s < 4; ++s) {
      v16h b = load_frag_b(Bt + s * 16 * K + k0, K, lane);
      acc[s] = wmma_f16(a, b, acc[s]);
    }
  }
}

// ---------------- converts / transposes -------------------------------------

__global__ void k_cvt(const float* s, _Float16* d, int n) {
  int i = blockIdx.x * 256 + threadIdx.x;
  if (i < n) d[i] = (_Float16)s[i];
}

// src (R,C) row-major f32 -> dst (C,R) row-major f16
__global__ void k_transpose_cvt(const float* s, _Float16* d, int R, int C) {
  int i = blockIdx.x * 256 + threadIdx.x;
  if (i >= R * C) return;
  int r = i / C, c = i - r * C;
  d[c * R + r] = (_Float16)s[i];
}

// ---------------- QKV projection --------------------------------------------
// Y = xh @ qkv_w + b (6144 x 768); 16x64 tile per wave. Writes Q,K as (N,256)
// f16 row-major; V transposed per head-dim: vt[(col-512)*NT + row].
__global__ void k_qkv(const _Float16* __restrict__ xh,
                      const _Float16* __restrict__ wt,
                      const float* __restrict__ bias,
                      _Float16* __restrict__ qh, _Float16* __restrict__ kh,
                      _Float16* __restrict__ vt) {
  const int ngroups = (3 * DM) / 64;               // 12 groups of 64 cols
  int wave = threadIdx.x >> 5, lane = threadIdx.x & 31;
  int tile = blockIdx.x * 8 + wave;
  if (tile >= (NT / 16) * ngroups) return;
  int mt = tile / ngroups, g = tile - mt * ngroups;
  v8f acc[4] = {};
  gemm_core4(xh + mt * 16 * DM, wt + g * 64 * DM, DM, lane, acc);
  int nloc = lane & 15, mb = (lane >> 4) * 8;
#pragma unroll
  for (int s = 0; s < 4; ++s) {
    int n = g * 64 + s * 16 + nloc;
    float bv = bias[n];
#pragma unroll
    for (int r = 0; r < 8; ++r) {
      int m = mt * 16 + mb + r;
      _Float16 hv = (_Float16)(acc[s][r] + bv);
      if (n < DM)            qh[m * DM + n] = hv;
      else if (n < 2 * DM)   kh[m * DM + (n - DM)] = hv;
      else                   vt[(n - 2 * DM) * NT + m] = hv;
    }
  }
}

// ---------------- generic WMMA GEMM with epilogues ---------------------------
// 16x64 tile per wave.
// EPI==1: exact GELU -> f16 outh.   EPI==2: + bias + resid -> f32 outf.
template <int EPI>
__global__ void k_gemm(const _Float16* __restrict__ A,
                       const _Float16* __restrict__ Bt,
                       const float* __restrict__ bias,
                       const float* __restrict__ resid,
                       float* __restrict__ outf, _Float16* __restrict__ outh,
                       int M, int Nn, int K) {
  int wave = threadIdx.x >> 5, lane = threadIdx.x & 31;
  int ngroups = Nn >> 6;                            // Nn divisible by 64
  int tile = blockIdx.x * 8 + wave;
  if (tile >= (M >> 4) * ngroups) return;
  int mt = tile / ngroups, g = tile - mt * ngroups;
  v8f acc[4] = {};
  gemm_core4(A + mt * 16 * K, Bt + g * 64 * K, K, lane, acc);
  int nloc = lane & 15, mb = (lane >> 4) * 8;
#pragma unroll
  for (int s = 0; s < 4; ++s) {
    int n = g * 64 + s * 16 + nloc;
    float bv = bias[n];
#pragma unroll
    for (int r = 0; r < 8; ++r) {
      int m = mt * 16 + mb + r;
      float v = acc[s][r] + bv;
      if constexpr (EPI == 1) {
        float gg = 0.5f * v * (1.0f + erff(v * 0.70710678118654752f));
        outh[m * Nn + n] = (_Float16)gg;
      } else {
        outf[m * Nn + n] = v + resid[m * Nn + n];
      }
    }
  }
}

// ---------------- doc-masked flash attention --------------------------------
// One block per 16-query tile; wave w = head w. Sorted doc ids -> key range
// via binary search; 32-key steps with online softmax; P -> A fragment via LDS.
__global__ void __launch_bounds__(128)
k_attn(const _Float16* __restrict__ qh, const _Float16* __restrict__ kh,
       const _Float16* __restrict__ vt, const int* __restrict__ doc,
       _Float16* __restrict__ oh) {
  __shared__ __align__(16) _Float16 pbuf[NH][16 * 32];
  int h = threadIdx.x >> 5, lane = threadIdx.x & 31;
  int q0 = blockIdx.x * 16;
  int nloc = lane & 15, mb = (lane >> 4) * 8;

  // doc-id range of this query tile (identical across the 4 head-waves)
  int dmy = doc[q0 + nloc];
  int dmin = dmy, dmax = dmy;
#pragma unroll
  for (int off = 1; off < 16; off <<= 1) {
    dmin = min(dmin, __shfl_xor(dmin, off, 32));
    dmax = max(dmax, __shfl_xor(dmax, off, 32));
  }
  int lo = 0, hi = NT;
  while (lo < hi) { int mid = (lo + hi) >> 1; if (doc[mid] <  dmin) lo = mid + 1; else hi = mid; }
  int kstart = lo;
  lo = 0; hi = NT;
  while (lo < hi) { int mid = (lo + hi) >> 1; if (doc[mid] <= dmax) lo = mid + 1; else hi = mid; }
  int kend = lo;

  const _Float16* qp = qh + q0 * DM + h * HD;
  v16h a0 = load_frag_a(qp, DM, lane);          // head-dim K = 0..31
  v16h a1 = load_frag_a(qp + 32, DM, lane);     // head-dim K = 32..63

  int qd[8];
#pragma unroll
  for (int r = 0; r < 8; ++r) qd[r] = doc[q0 + mb + r];

  float mrow[8], lrow[8];
#pragma unroll
  for (int r = 0; r < 8; ++r) { mrow[r] = -__builtin_inff(); lrow[r] = 0.f; }
  v8f oacc[4] = {};

  int t0 = kstart >> 5, t1 = (kend + 31) >> 5;  // same trip count for all waves
  for (int t = t0; t < t1; ++t) {
    int kb0 = t << 5;
    // S = Q K^T for two 16-key sub-tiles (keys are contiguous rows of kh)
    v8f s0 = {}, s1 = {};
    {
      const _Float16* kp = kh + kb0 * DM + h * HD;
      s0 = wmma_f16(a0, load_frag_b(kp, DM, lane), s0);
      s0 = wmma_f16(a1, load_frag_b(kp + 32, DM, lane), s0);
      const _Float16* kp2 = kp + 16 * DM;
      s1 = wmma_f16(a0, load_frag_b(kp2, DM, lane), s1);
      s1 = wmma_f16(a1, load_frag_b(kp2 + 32, DM, lane), s1);
    }
    int kd0 = doc[kb0 + nloc];
    int kd1 = doc[kb0 + 16 + nloc];
#pragma unroll
    for (int r = 0; r < 8; ++r) {
      bool m0 = (kd0 == qd[r]);
      bool m1 = (kd1 == qd[r]);
      float v0 = s0[r] * 0.125f;                 // 1/sqrt(64)
      float v1 = s1[r] * 0.125f;
      float rm = fmaxf(m0 ? v0 : -__builtin_inff(), m1 ? v1 : -__builtin_inff());
#pragma unroll
      for (int off = 1; off < 16; off <<= 1) rm = fmaxf(rm, __shfl_xor(rm, off, 32));
      float mnew = fmaxf(mrow[r], rm);
      float corr = (mnew < -1e37f) ? 0.f : __expf(mrow[r] - mnew);
      float p0 = m0 ? __expf(v0 - mnew) : 0.f;
      float p1 = m1 ? __expf(v1 - mnew) : 0.f;
      float ps = p0 + p1;
#pragma unroll
      for (int off = 1; off < 16; off <<= 1) ps += __shfl_xor(ps, off, 32);
      lrow[r] = lrow[r] * corr + ps;
      mrow[r] = mnew;
#pragma unroll
      for (int dc = 0; dc < 4; ++dc) oacc[dc][r] *= corr;
      pbuf[h][(mb + r) * 32 + nloc]      = (_Float16)p0;   // C-layout -> row-major
      pbuf[h][(mb + r) * 32 + 16 + nloc] = (_Float16)p1;
    }
    __syncthreads();                              // uniform across all 4 waves
    v16h ap = load_frag_a(&pbuf[h][0], 32, lane); // P as 16x32 A fragment
    __syncthreads();
#pragma unroll
    for (int dc = 0; dc < 4; ++dc) {
      // V^T (hd-major) gives contiguous keys per lane for the B fragment
      v16h bv = load_frag_b(vt + (h * HD + dc * 16) * NT + kb0, NT, lane);
      oacc[dc] = wmma_f16(ap, bv, oacc[dc]);
    }
  }
#pragma unroll
  for (int r = 0; r < 8; ++r) {
    float inv = 1.0f / lrow[r];
    int m = q0 + mb + r;
#pragma unroll
    for (int dc = 0; dc < 4; ++dc)
      oh[m * DM + h * HD + dc * 16 + nloc] = (_Float16)(oacc[dc][r] * inv);
  }
}

// ---------------- LayerNorm (wave per row, D=256) ---------------------------
__global__ void k_ln(const float* __restrict__ in, const float* __restrict__ g,
                     const float* __restrict__ b, float* __restrict__ outf,
                     _Float16* __restrict__ outh, int rows) {
  int wave = threadIdx.x >> 5, lane = threadIdx.x & 31;
  int row = blockIdx.x * 8 + wave;
  if (row >= rows) return;
  const float* p = in + row * DM;
  float v[8]; float s = 0.f;
#pragma unroll
  for (int j = 0; j < 8; ++j) { v[j] = p[lane + j * 32]; s += v[j]; }
#pragma unroll
  for (int off = 16; off > 0; off >>= 1) s += __shfl_xor(s, off, 32);
  float mu = s * (1.0f / 256.0f);
  float q = 0.f;
#pragma unroll
  for (int j = 0; j < 8; ++j) { float d = v[j] - mu; q += d * d; }
#pragma unroll
  for (int off = 16; off > 0; off >>= 1) q += __shfl_xor(q, off, 32);
  float rstd = rsqrtf(q * (1.0f / 256.0f) + 1e-5f);
#pragma unroll
  for (int j = 0; j < 8; ++j) {
    int c = lane + j * 32;
    float o = (v[j] - mu) * rstd * g[c] + b[c];
    outf[row * DM + c] = o;
    if (outh) outh[row * DM + c] = (_Float16)o;
  }
}

// ---------------- host side --------------------------------------------------
extern "C" void kernel_launch(void* const* d_in, const int* in_sizes, int n_in,
                              void* d_out, int out_size, void* d_ws, size_t ws_size,
                              hipStream_t stream) {
  (void)in_sizes; (void)n_in; (void)out_size; (void)ws_size;
  const float* x     = (const float*)d_in[0];
  const int*   doc   = (const int*)d_in[1];
  const float* qkv_w = (const float*)d_in[2];
  const float* qkv_b = (const float*)d_in[3];
  const float* out_w = (const float*)d_in[4];
  const float* out_b = (const float*)d_in[5];
  const float* ff_w1 = (const float*)d_in[6];
  const float* ff_b1 = (const float*)d_in[7];
  const float* ff_w2 = (const float*)d_in[8];
  const float* ff_b2 = (const float*)d_in[9];
  const float* ln1_g = (const float*)d_in[10];
  const float* ln1_b = (const float*)d_in[11];
  const float* ln2_g = (const float*)d_in[12];
  const float* ln2_b = (const float*)d_in[13];
  float* out = (float*)d_out;

  char* ws = (char*)d_ws;
  size_t off = 0;
  auto alloc = [&](size_t bytes) -> void* {
    void* p = ws + off;
    off += (bytes + 255) & ~(size_t)255;
    return p;
  };
  _Float16* xh     = (_Float16*)alloc((size_t)NT * DM * 2);
  _Float16* qkv_wt = (_Float16*)alloc((size_t)3 * DM * DM * 2);
  _Float16* out_wt = (_Float16*)alloc((size_t)DM * DM * 2);
  _Float16* ffw1t  = (_Float16*)alloc((size_t)DFF * DM * 2);
  _Float16* ffw2t  = (_Float16*)alloc((size_t)DM * DFF * 2);
  _Float16* qh     = (_Float16*)alloc((size_t)NT * DM * 2);
  _Float16* kh     = (_Float16*)alloc((size_t)NT * DM * 2);
  _Float16* vt     = (_Float16*)alloc((size_t)NT * DM * 2);
  _Float16* oh     = (_Float16*)alloc((size_t)NT * DM * 2);
  float*    y1     = (float*)alloc((size_t)NT * DM * 4);
  float*    xn1    = (float*)alloc((size_t)NT * DM * 4);
  _Float16* xn1h   = (_Float16*)alloc((size_t)NT * DM * 2);
  _Float16* hff    = (_Float16*)alloc((size_t)NT * DFF * 2);
  float*    y2     = (float*)alloc((size_t)NT * DM * 4);

  // precision converts + weight transposes (weights stored N-major for B frags)
  k_cvt<<<(NT * DM + 255) / 256, 256, 0, stream>>>(x, xh, NT * DM);
  k_transpose_cvt<<<(DM * 3 * DM + 255) / 256, 256, 0, stream>>>(qkv_w, qkv_wt, DM, 3 * DM);
  k_transpose_cvt<<<(DM * DM + 255) / 256, 256, 0, stream>>>(out_w, out_wt, DM, DM);
  k_transpose_cvt<<<(DM * DFF + 255) / 256, 256, 0, stream>>>(ff_w1, ffw1t, DM, DFF);
  k_transpose_cvt<<<(DFF * DM + 255) / 256, 256, 0, stream>>>(ff_w2, ffw2t, DFF, DM);

  { int tiles = (NT / 16) * ((3 * DM) / 64);
    k_qkv<<<(tiles + 7) / 8, 256, 0, stream>>>(xh, qkv_wt, qkv_b, qh, kh, vt); }

  k_attn<<<NT / 16, 128, 0, stream>>>(qh, kh, vt, doc, oh);

  { int tiles = (NT / 16) * (DM / 64);
    k_gemm<2><<<(tiles + 7) / 8, 256, 0, stream>>>(oh, out_wt, out_b, x, y1, nullptr,
                                                   NT, DM, DM); }
  k_ln<<<NT / 8, 256, 0, stream>>>(y1, ln1_g, ln1_b, xn1, xn1h, NT);

  { int tiles = (NT / 16) * (DFF / 64);
    k_gemm<1><<<(tiles + 7) / 8, 256, 0, stream>>>(xn1h, ffw1t, ff_b1, nullptr, nullptr,
                                                   hff, NT, DFF, DM); }
  { int tiles = (NT / 16) * (DM / 64);
    k_gemm<2><<<(tiles + 7) / 8, 256, 0, stream>>>(hff, ffw2t, ff_b2, xn1, y2, nullptr,
                                                   NT, DM, DFF); }
  k_ln<<<NT / 8, 256, 0, stream>>>(y2, ln2_g, ln2_b, out, nullptr, NT);
}
